// L2_78769700208929
// MI455X (gfx1250) — compile-verified, matrix operand order checked by reading, not run on hardware
//
#include <hip/hip_runtime.h>

typedef __attribute__((ext_vector_type(16))) _Float16 v16h;
typedef __attribute__((ext_vector_type(8)))  _Float16 v8h;
typedef __attribute__((ext_vector_type(4)))  _Float16 v4h;
typedef __attribute__((ext_vector_type(8)))  float    v8f;
typedef __attribute__((ext_vector_type(4)))  float    v4f;

#define DIM    1024      // K (feature dim)
#define NB     4096      // rows of x
#define NN     8192      // rows of y
#define TILE   128       // output tile edge per block
#define KT     32        // K step (one f16 WMMA K)
#define LDST   40        // LDS row stride in f16 (32 + 8 pad -> 80B, bank-conflict free)

// ---------------------------------------------------------------------------
// Pre-pass: exact f32 squared row norms for x (4096 rows) and y (8192 rows).
// One block per row, 256 threads. Results into workspace.
// ---------------------------------------------------------------------------
__global__ void l2_row_norms(const float* __restrict__ x,
                             const float* __restrict__ y,
                             float* __restrict__ xs,
                             float* __restrict__ ys) {
    __shared__ float red[8];
    const int row = blockIdx.x;
    const float* src;
    float* dst;
    if (row < NB) { src = x + (size_t)row * DIM;        dst = xs + row;        }
    else          { src = y + (size_t)(row - NB) * DIM; dst = ys + (row - NB); }

    float s = 0.0f;
    for (int i = threadIdx.x; i < DIM; i += 256) {
        float v = src[i];
        s += v * v;
    }
    // wave32 reduction
    #pragma unroll
    for (int off = 16; off > 0; off >>= 1)
        s += __shfl_down(s, off, 32);
    if ((threadIdx.x & 31) == 0) red[threadIdx.x >> 5] = s;
    __syncthreads();
    if (threadIdx.x == 0) {
        float t = 0.0f;
        #pragma unroll
        for (int w = 0; w < 8; ++w) t += red[w];
        *dst = t;
    }
}

// ---------------------------------------------------------------------------
// Main kernel: out[b,n] = 2 * (x . y) - ||x||^2 - ||y||^2
// 128x128 tile per 256-thread block; f16 WMMA 16x16x32 with f32 accum.
// ---------------------------------------------------------------------------
__device__ __forceinline__ void load_ktile(const float* __restrict__ xp,
                                           const float* __restrict__ yp,
                                           v4f ax[4], v4f by[4]) {
    #pragma unroll
    for (int p = 0; p < 4; ++p) {
        ax[p] = *(const v4f*)(xp + (size_t)(32 * p) * DIM);
        by[p] = *(const v4f*)(yp + (size_t)(32 * p) * DIM);
    }
}

__global__ __launch_bounds__(256, 1)
void l2_wmma_f16(const float* __restrict__ x, const float* __restrict__ y,
                 const float* __restrict__ xs, const float* __restrict__ ys,
                 float* __restrict__ out) {
    __shared__ __align__(16) _Float16 As[2][TILE][LDST];
    __shared__ __align__(16) _Float16 Bs[2][TILE][LDST];
    __shared__ float xs_s[TILE];
    __shared__ float ys_s[TILE];

    const int tid     = threadIdx.x;
    const int rowBase = blockIdx.y * TILE;   // into x (B dim)
    const int colBase = blockIdx.x * TILE;   // into y (N dim)

    // stage norms for this tile (synchronized by first barrier in the loop)
    if (tid < TILE) xs_s[tid]        = xs[rowBase + tid];
    else            ys_s[tid - TILE] = ys[colBase + (tid - TILE)];

    // global->LDS loader mapping: 8 float4 per row of 32, 32 rows per pass, 4 passes
    const int lc = tid & 7;                  // float4 column within K-tile
    const int lr = tid >> 3;                 // row 0..31
    const float* xsrc = x + (size_t)(rowBase + lr) * DIM + lc * 4;
    const float* ysrc = y + (size_t)(colBase + lr) * DIM + lc * 4;

    // wave / fragment mapping: 8 waves in 2x4 grid, each wave owns 64x32
    const int lane = tid & 31;
    const int wave = tid >> 5;
    const int wr   = wave >> 2;              // 0..1
    const int wc   = wave & 3;               // 0..3
    const int r0   = wr * 64;
    const int c0   = wc * 32;
    const int m16  = lane & 15;              // row/col within 16x16 fragment
    const int hk   = (lane >> 4) << 3;       // 0 or 8 : K sub-offset per A/B layout
    const int rsub = (lane >> 4) << 3;       // 0 or 8 : M sub-offset in C/D layout

    v8f acc[4][2] = {};

    v4f ax[4], by[4], axn[4], byn[4];
    load_ktile(xsrc, ysrc, ax, by);

    const int NT = DIM / KT;                 // 32 K-steps
    for (int kt = 0; kt < NT; ++kt) {
        const int buf = kt & 1;

        // convert f32 -> f16 and stage into LDS
        #pragma unroll
        for (int p = 0; p < 4; ++p) {
            *(v4h*)&As[buf][lr + 32 * p][lc * 4] = __builtin_convertvector(ax[p], v4h);
            *(v4h*)&Bs[buf][lr + 32 * p][lc * 4] = __builtin_convertvector(by[p], v4h);
        }

        // prefetch next K tile into registers (overlaps WMMA below)
        if (kt + 1 < NT)
            load_ktile(xsrc + (kt + 1) * KT, ysrc + (kt + 1) * KT, axn, byn);
        // L2 prefetch two tiles ahead
        if (kt + 2 < NT) {
            __builtin_prefetch(xsrc + (kt + 2) * KT, 0, 0);
            __builtin_prefetch(ysrc + (kt + 2) * KT, 0, 0);
        }

        __syncthreads();

        // load fragments from LDS (A-style layout serves both A and B=y^T)
        v16h a[4], b[2];
        #pragma unroll
        for (int i = 0; i < 4; ++i) {
            const _Float16* pa = &As[buf][r0 + 16 * i + m16][hk];
            v8h c0v = *(const v8h*)pa;
            v8h c1v = *(const v8h*)(pa + 16);
            a[i] = __builtin_shufflevector(c0v, c1v,
                                           0, 1, 2, 3, 4, 5, 6, 7,
                                           8, 9, 10, 11, 12, 13, 14, 15);
        }
        #pragma unroll
        for (int j = 0; j < 2; ++j) {
            const _Float16* pb = &Bs[buf][c0 + 16 * j + m16][hk];
            v8h c0v = *(const v8h*)pb;
            v8h c1v = *(const v8h*)(pb + 16);
            b[j] = __builtin_shufflevector(c0v, c1v,
                                           0, 1, 2, 3, 4, 5, 6, 7,
                                           8, 9, 10, 11, 12, 13, 14, 15);
        }

        #pragma unroll
        for (int i = 0; i < 4; ++i)
            #pragma unroll
            for (int j = 0; j < 2; ++j)
                acc[i][j] = __builtin_amdgcn_wmma_f32_16x16x32_f16(
                    /*neg_a=*/false, a[i], /*neg_b=*/false, b[j],
                    /*c_mod=*/(short)0, acc[i][j],
                    /*reuse_a=*/false, /*reuse_b=*/false);

        #pragma unroll
        for (int p = 0; p < 4; ++p) { ax[p] = axn[p]; by[p] = byn[p]; }
    }

    // epilogue: out = 2*dot - ||x||^2 - ||y||^2
    #pragma unroll
    for (int i = 0; i < 4; ++i) {
        const int trow = r0 + 16 * i + rsub;
        #pragma unroll
        for (int j = 0; j < 2; ++j) {
            const int tcol = c0 + 16 * j + m16;
            const float yn = ys_s[tcol];
            float* op = out + (size_t)(rowBase + trow) * NN + (colBase + tcol);
            v8f c = acc[i][j];
            #pragma unroll
            for (int v = 0; v < 8; ++v)
                op[(size_t)v * NN] = 2.0f * c[v] - xs_s[trow + v] - yn;
        }
    }
}

// ---------------------------------------------------------------------------
extern "C" void kernel_launch(void* const* d_in, const int* in_sizes, int n_in,
                              void* d_out, int out_size, void* d_ws, size_t ws_size,
                              hipStream_t stream) {
    const float* x = (const float*)d_in[0];   // [4096, 1024]
    const float* y = (const float*)d_in[1];   // [8192, 1024]
    float* out = (float*)d_out;               // [4096, 8192]

    float* xs = (float*)d_ws;                 // 4096 floats
    float* ys = xs + NB;                      // 8192 floats (48 KB total)

    l2_row_norms<<<NB + NN, 256, 0, stream>>>(x, y, xs, ys);

    dim3 grid(NN / TILE, NB / TILE);          // (64, 32)
    l2_wmma_f16<<<grid, 256, 0, stream>>>(x, y, xs, ys, out);
}
